// TAttention_16793322127945
// MI455X (gfx1250) — matrix operand order repaired and checked by reading, not compiled, not run on hardware
//
#include <hip/hip_runtime.h>

typedef _Float16 half_t;
typedef __attribute__((ext_vector_type(16))) _Float16 v16h;
typedef __attribute__((ext_vector_type(8)))  _Float16 v8h;
typedef __attribute__((ext_vector_type(8)))  float    v8f;

#define D_MODEL 256
#define NHEAD   4
#define HEAD_DIM 64
#define SEQ     4096
#define BATCH   4
#define NTOK    (BATCH*SEQ)
#define AW      4   // waves per block in attention kernel

static __device__ __forceinline__ v8f wmma_f16(v16h a, v16h b, v8f c) {
  return __builtin_amdgcn_wmma_f32_16x16x32_f16(false, a, false, b, (short)0, c, false, false);
}

// A-fragment (16x32 f16): lane L holds row (L&15); halves 0..7 -> K = (L>>4)*8 + 0..7,
// halves 8..15 -> K = 16 + (L>>4)*8 + 0..7   (ISA 7.12.2)
static __device__ __forceinline__ v16h load_A_g(const half_t* p, int ld, int row0, int k0, int lane) {
  const half_t* s = p + (long)(row0 + (lane & 15)) * ld + k0 + ((lane >> 4) << 3);
  v8h lo = *(const v8h*)(s);
  v8h hi = *(const v8h*)(s + 16);
  v16h a;
#pragma unroll
  for (int i = 0; i < 8; ++i) { a[i] = lo[i]; a[i + 8] = hi[i]; }
  return a;
}

// B-fragment (32x16 f16) of B[k][n] = M[n*ld + k] (row-major M, i.e. x @ M^T):
// lane L holds col (L&15); halves 0..7 -> K = (L>>4)*8 + 0..7, halves 8..15 -> K = 16 + (L>>4)*8 + 0..7.
static __device__ __forceinline__ v16h load_B_g(const half_t* p, int ld, int n0, int k0, int lane) {
  const half_t* s = p + (long)(n0 + (lane & 15)) * ld + k0 + ((lane >> 4) << 3);
  v8h lo = *(const v8h*)(s);
  v8h hi = *(const v8h*)(s + 16);
  v16h b;
#pragma unroll
  for (int i = 0; i < 8; ++i) { b[i] = lo[i]; b[i + 8] = hi[i]; }
  return b;
}

// P-tile (16 rows x 64 cols f16 in LDS, row stride 64) -> A-fragment at K offset k0
static __device__ __forceinline__ v16h load_A_lds(const half_t* p, int k0, int lane) {
  const half_t* s = p + (lane & 15) * 64 + k0 + ((lane >> 4) << 3);
  v8h lo = *(const v8h*)(s);
  v8h hi = *(const v8h*)(s + 16);
  v16h a;
#pragma unroll
  for (int i = 0; i < 8; ++i) { a[i] = lo[i]; a[i + 8] = hi[i]; }
  return a;
}

__global__ void k_cvt(const float* __restrict__ src, half_t* __restrict__ dst, int n) {
  int i = blockIdx.x * blockDim.x + threadIdx.x;
  if (i < n) dst[i] = (half_t)src[i];
}

// LayerNorm over 256-wide rows: one wave32 per row, 8 elements per lane.
__global__ void __launch_bounds__(256) k_ln(const float* __restrict__ x, const float* __restrict__ g,
                                            const float* __restrict__ b, half_t* __restrict__ out, int nrows) {
  int row = blockIdx.x * (blockDim.x >> 5) + (threadIdx.x >> 5);
  if (row >= nrows) return;
  int lane = threadIdx.x & 31;
  const float* xr = x + (long)row * D_MODEL;
  float v[8];
  float s = 0.f;
#pragma unroll
  for (int i = 0; i < 8; ++i) { v[i] = xr[i * 32 + lane]; s += v[i]; }
#pragma unroll
  for (int o = 16; o > 0; o >>= 1) s += __shfl_xor(s, o, 32);
  float mu = s * (1.f / D_MODEL);
  float vs = 0.f;
#pragma unroll
  for (int i = 0; i < 8; ++i) { float d = v[i] - mu; vs += d * d; }
#pragma unroll
  for (int o = 16; o > 0; o >>= 1) vs += __shfl_xor(vs, o, 32);
  float rs = rsqrtf(vs * (1.f / D_MODEL) + 1e-5f);
#pragma unroll
  for (int i = 0; i < 8; ++i) {
    int c = i * 32 + lane;
    out[(long)row * D_MODEL + c] = (half_t)((v[i] - mu) * rs * g[c] + b[c]);
  }
}

// QKV projection: per-wave 16x64 tile = one head slice of one of {Q,K,V}.
// All 4 B-fragments of a k-step are loaded before the WMMAs (wide clause, staged waits).
// Q gets 1/TEMP=0.125 folded in. Q,K stored [b,h,s,hd]; V stored transposed [b,h,hd,s].
__global__ void __launch_bounds__(256) k_qkv(const half_t* __restrict__ xn,
                                             const half_t* __restrict__ wq, const half_t* __restrict__ wk,
                                             const half_t* __restrict__ wv,
                                             half_t* __restrict__ q, half_t* __restrict__ k,
                                             half_t* __restrict__ vT) {
  int wave = blockIdx.x * (blockDim.x >> 5) + (threadIdx.x >> 5);
  int lane = threadIdx.x & 31;
  int rt = wave / 12;
  int rem = wave % 12;
  int m = rem >> 2;  // 0=Q 1=K 2=V
  int h = rem & 3;   // head (64-wide col tile)
  if (rt >= NTOK / 16) return;
  const half_t* W = (m == 0) ? wq : (m == 1) ? wk : wv;
  v8f acc[4] = {};
#pragma unroll
  for (int kt = 0; kt < 8; ++kt) {
    v16h a = load_A_g(xn, D_MODEL, rt * 16, kt * 32, lane);
    v16h b[4];
#pragma unroll
    for (int n = 0; n < 4; ++n)
      b[n] = load_B_g(W, D_MODEL, h * HEAD_DIM + n * 16, kt * 32, lane);
#pragma unroll
    for (int n = 0; n < 4; ++n)
      acc[n] = wmma_f16(a, b[n], acc[n]);
  }
  float scale = (m == 0) ? 0.125f : 1.0f;
#pragma unroll
  for (int n = 0; n < 4; ++n) {
    int d = n * 16 + (lane & 15);  // dim within head
#pragma unroll
    for (int i = 0; i < 8; ++i) {
      int row = rt * 16 + ((lane >> 4) << 3) + i;  // = b*SEQ + s
      int bb = row >> 12;
      int ss = row & (SEQ - 1);
      half_t val = (half_t)(acc[n][i] * scale);
      long bh = (long)(bb * NHEAD + h);
      if (m == 2)      vT[(bh * HEAD_DIM + d) * SEQ + ss] = val;
      else if (m == 1) k[(bh * SEQ + ss) * HEAD_DIM + d] = val;
      else             q[(bh * SEQ + ss) * HEAD_DIM + d] = val;
    }
  }
}

// Flash attention: one wave per 16-query tile, keys in blocks of 64, online softmax
// (cross-lane reductions amortized over 64 keys), fused residual: xt = x + attn_out.
// All K-fragments (resp. V-fragments) loaded before their WMMA group.
__global__ void __launch_bounds__(32 * AW) k_attn(const half_t* __restrict__ q, const half_t* __restrict__ k,
                                                  const half_t* __restrict__ vT,
                                                  const float* __restrict__ x, float* __restrict__ xt) {
  __shared__ half_t plds[AW][16 * 64];
  int wid = threadIdx.x >> 5;
  int lane = threadIdx.x & 31;
  int wave = blockIdx.x * AW + wid;
  int qt = wave & (SEQ / 16 - 1);
  int bh = wave >> 8;
  if (bh >= BATCH * NHEAD) return;
  const half_t* qp = q + (long)bh * SEQ * HEAD_DIM;
  const half_t* kp = k + (long)bh * SEQ * HEAD_DIM;
  const half_t* vp = vT + (long)bh * HEAD_DIM * SEQ;
  v16h aq0 = load_A_g(qp, HEAD_DIM, qt * 16, 0, lane);
  v16h aq1 = load_A_g(qp, HEAD_DIM, qt * 16, 32, lane);
  v8f o[4] = {};
  float mrow[8], lrow[8];
#pragma unroll
  for (int i = 0; i < 8; ++i) { mrow[i] = -1e30f; lrow[i] = 0.f; }
  half_t* P = plds[wid];
  int col = lane & 15;
  int nkb = ((qt * 16 + 16) + 63) >> 6;
  for (int kb = 0; kb < nkb; ++kb) {
    int jb = kb * 64;
    // S = Q K^T : 4 col-tiles x (inner dim 64 -> 2 wmma each); batch the 8 loads first
    v16h bk[4][2];
#pragma unroll
    for (int t = 0; t < 4; ++t) {
      bk[t][0] = load_B_g(kp, HEAD_DIM, jb + t * 16, 0, lane);
      bk[t][1] = load_B_g(kp, HEAD_DIM, jb + t * 16, 32, lane);
    }
    v8f s[4];
#pragma unroll
    for (int t = 0; t < 4; ++t) {
      v8f z = {};
      z = wmma_f16(aq0, bk[t][0], z);
      s[t] = wmma_f16(aq1, bk[t][1], z);
    }
    if (jb + 63 > qt * 16) {  // causal mask only on the diagonal-straddling block
#pragma unroll
      for (int t = 0; t < 4; ++t) {
#pragma unroll
        for (int i = 0; i < 8; ++i) {
          int ri = qt * 16 + ((lane >> 4) << 3) + i;
          if (jb + t * 16 + col > ri) s[t][i] = -1e30f;
        }
      }
    }
    // online softmax: one max-reduce + one sum-reduce per 64 keys
    float a[8];
#pragma unroll
    for (int i = 0; i < 8; ++i) {
      float mx = fmaxf(fmaxf(s[0][i], s[1][i]), fmaxf(s[2][i], s[3][i]));
#pragma unroll
      for (int o2 = 8; o2 > 0; o2 >>= 1) mx = fmaxf(mx, __shfl_xor(mx, o2, 32));
      float mn = fmaxf(mrow[i], mx);
      a[i] = __expf(mrow[i] - mn);
      mrow[i] = mn;
      float sm = 0.f;
#pragma unroll
      for (int t = 0; t < 4; ++t) {
        float p = __expf(s[t][i] - mn);
        s[t][i] = p;
        sm += p;
      }
#pragma unroll
      for (int o2 = 8; o2 > 0; o2 >>= 1) sm += __shfl_xor(sm, o2, 32);
      lrow[i] = lrow[i] * a[i] + sm;
    }
#pragma unroll
    for (int dt = 0; dt < 4; ++dt)
#pragma unroll
      for (int i = 0; i < 8; ++i) o[dt][i] *= a[i];
    // C-layout f32 -> A-layout f16 via private LDS tile (per-wave; LDS is in-order)
#pragma unroll
    for (int t = 0; t < 4; ++t)
#pragma unroll
      for (int i = 0; i < 8; ++i) {
        int r = ((lane >> 4) << 3) + i;
        P[r * 64 + t * 16 + col] = (half_t)s[t][i];
      }
    // O += P V : batch the 8 V loads, then 8 wmmas
    v16h bv[4][2];
#pragma unroll
    for (int dt = 0; dt < 4; ++dt) {
      bv[dt][0] = load_B_g(vp, SEQ, dt * 16, jb, lane);
      bv[dt][1] = load_B_g(vp, SEQ, dt * 16, jb + 32, lane);
    }
    v16h pa0 = load_A_lds(P, 0, lane);
    v16h pa1 = load_A_lds(P, 32, lane);
#pragma unroll
    for (int dt = 0; dt < 4; ++dt) {
      o[dt] = wmma_f16(pa0, bv[dt][0], o[dt]);
      o[dt] = wmma_f16(pa1, bv[dt][1], o[dt]);
    }
  }
  int b = bh >> 2, h = bh & 3;
#pragma unroll
  for (int i = 0; i < 8; ++i) {
    float inv = 1.f / lrow[i];
    int s = qt * 16 + ((lane >> 4) << 3) + i;
    long base = ((long)(b * SEQ + s)) * D_MODEL + h * HEAD_DIM;
#pragma unroll
    for (int dt = 0; dt < 4; ++dt)
      xt[base + dt * 16 + col] = x[base + dt * 16 + col] + o[dt][i] * inv;
  }
}

// Generic 256x256 GEMM (y = X @ W^T + bias), per-wave 16x64 tile (A reused over 4 WMMAs),
// batched B loads per k-step, optional ReLU->f16 out, or residual+f32 out.
__global__ void __launch_bounds__(256) k_ffn(const half_t* __restrict__ X, const half_t* __restrict__ W,
                                             const float* __restrict__ bias, const float* __restrict__ resid,
                                             half_t* __restrict__ outH, float* __restrict__ outF) {
  int wave = blockIdx.x * (blockDim.x >> 5) + (threadIdx.x >> 5);
  int lane = threadIdx.x & 31;
  int rt = wave >> 2;
  int ct = wave & 3;  // 64-wide col tile
  if (rt >= NTOK / 16) return;
  v8f acc[4] = {};
#pragma unroll
  for (int kt = 0; kt < 8; ++kt) {
    v16h a = load_A_g(X, D_MODEL, rt * 16, kt * 32, lane);
    v16h b[4];
#pragma unroll
    for (int n = 0; n < 4; ++n)
      b[n] = load_B_g(W, D_MODEL, ct * 64 + n * 16, kt * 32, lane);
#pragma unroll
    for (int n = 0; n < 4; ++n)
      acc[n] = wmma_f16(a, b[n], acc[n]);
  }
#pragma unroll
  for (int n = 0; n < 4; ++n) {
    int colg = ct * 64 + n * 16 + (lane & 15);
#pragma unroll
    for (int i = 0; i < 8; ++i) {
      long row = rt * 16 + ((lane >> 4) << 3) + i;
      float v = acc[n][i] + bias[colg];
      if (outH) {
        outH[row * D_MODEL + colg] = (half_t)fmaxf(v, 0.f);
      } else {
        outF[row * D_MODEL + colg] = v + resid[row * D_MODEL + colg];
      }
    }
  }
}

extern "C" void kernel_launch(void* const* d_in, const int* in_sizes, int n_in,
                              void* d_out, int out_size, void* d_ws, size_t ws_size,
                              hipStream_t stream) {
  const float* x   = (const float*)d_in[0];
  const float* Wq  = (const float*)d_in[1];
  const float* Wk  = (const float*)d_in[2];
  const float* Wv  = (const float*)d_in[3];
  const float* g1  = (const float*)d_in[4];
  const float* b1  = (const float*)d_in[5];
  const float* g2  = (const float*)d_in[6];
  const float* b2  = (const float*)d_in[7];
  const float* W1  = (const float*)d_in[8];
  const float* bf1 = (const float*)d_in[9];
  const float* W2  = (const float*)d_in[10];
  const float* bf2 = (const float*)d_in[11];

  char* ws = (char*)d_ws;
  half_t* xn = (half_t*)(ws);                        // 8 MB
  half_t* q  = (half_t*)(ws + (8ull  << 20));        // 8 MB
  half_t* kk = (half_t*)(ws + (16ull << 20));        // 8 MB
  half_t* vT = (half_t*)(ws + (24ull << 20));        // 8 MB
  float*  xt = (float*) (ws + (32ull << 20));        // 16 MB
  half_t* hn = (half_t*)(ws + (48ull << 20));        // 8 MB
  half_t* h1 = (half_t*)(ws + (56ull << 20));        // 8 MB
  half_t* wh = (half_t*)(ws + (64ull << 20));        // 5 * 128 KB
  half_t* wqh = wh;
  half_t* wkh = wh + 1 * 65536;
  half_t* wvh = wh + 2 * 65536;
  half_t* w1h = wh + 3 * 65536;
  half_t* w2h = wh + 4 * 65536;

  const int WN = D_MODEL * D_MODEL;  // 65536
  k_cvt<<<WN / 256, 256, 0, stream>>>(Wq, wqh, WN);
  k_cvt<<<WN / 256, 256, 0, stream>>>(Wk, wkh, WN);
  k_cvt<<<WN / 256, 256, 0, stream>>>(Wv, wvh, WN);
  k_cvt<<<WN / 256, 256, 0, stream>>>(W1, w1h, WN);
  k_cvt<<<WN / 256, 256, 0, stream>>>(W2, w2h, WN);

  k_ln<<<NTOK / 8, 256, 0, stream>>>(x, g1, b1, xn, NTOK);

  k_qkv<<<(NTOK / 16 * 12) / 8, 256, 0, stream>>>(xn, wqh, wkh, wvh, q, kk, vT);

  k_attn<<<(BATCH * NHEAD * (SEQ / 16)) / AW, 32 * AW, 0, stream>>>(q, kk, vT, x, xt);

  k_ln<<<NTOK / 8, 256, 0, stream>>>(xt, g2, b2, hn, NTOK);

  k_ffn<<<(NTOK / 16 * 4) / 8, 256, 0, stream>>>(hn, w1h, bf1, nullptr, h1, nullptr);
  k_ffn<<<(NTOK / 16 * 4) / 8, 256, 0, stream>>>(h1, w2h, bf2, xt, nullptr, (float*)d_out);
}
